// SemanticsModulatedAttention_86414741996152
// MI455X (gfx1250) — matrix operand order repaired and checked by reading, not compiled, not run on hardware
//
#include <hip/hip_runtime.h>
#include <hip/hip_bf16.h>
#include <math.h>

#define Bn   32
#define Tn   1024
#define Dn   512
#define DTn  256
#define Hn   8
#define DHn  64
#define Rn   4
#define Mm   196
#define RMn  784
#define NTn  77
#define Nn   1885
#define NPAD 1888
#define En   2048
#define NEGV (-1000000.0f)

typedef __bf16 bf16_t;
typedef __attribute__((ext_vector_type(8)))  __bf16 v8bf;
typedef __attribute__((ext_vector_type(16))) __bf16 v16bf;
typedef __attribute__((ext_vector_type(8)))  float  v8f;
typedef __attribute__((ext_vector_type(4)))  unsigned int v4u;
typedef __attribute__((ext_vector_type(8)))  int v8i;
typedef __attribute__((ext_vector_type(4)))  int v4i;

union FragAB { v16bf v; v8bf p[2]; };

#if defined(__has_builtin)
#if __has_builtin(__builtin_amdgcn_tensor_load_to_lds)
#define USE_TDM 1
#endif
#endif

__device__ __forceinline__ float wave_sum(float v){
#pragma unroll
  for (int off = 16; off > 0; off >>= 1) v += __shfl_xor(v, off, 32);
  return v;
}
__device__ __forceinline__ float silu_f(float x){ return x / (1.0f + __expf(-x)); }

#ifdef USE_TDM
// Post a TDM descriptor: load a (rows<=128) x 32-half 2D tile (row stride K halfs)
// from global into LDS at byte offset lds_off, inserting 4 DWORDs of padding after
// every 16 DWORDs (=> 40-half LDS row stride). OOB rows (beyond rows_rem) read zero.
__device__ __forceinline__ void tdm_load_tile_a(const bf16_t* gsrc, unsigned lds_off,
                                                int K, int rows_rem){
  unsigned long long ga = (unsigned long long)(size_t)gsrc;
  v4u g0;
  g0[0] = 1u;                                        // count=1 valid descriptor
  g0[1] = lds_off;                                   // LDS byte address
  g0[2] = (unsigned)(ga & 0xFFFFFFFFu);              // global addr [95:64]
  g0[3] = (unsigned)((ga >> 32) & 0x01FFFFFFu) | (2u << 30);  // addr hi | type=2
  v8i g1;
  g1[0] = (1 << 16)        // data_size = 2 bytes
        | (1 << 20)        // pad_enable
        | (3 << 22)        // pad_interval: 16 DWORDs (32 halfs)
        | (3 << 25);       // pad_amount: 4 DWORDs (8 halfs)
  g1[1] = (int)(((unsigned)K & 0xFFFFu) << 16);                    // tensor_dim0 lo16
  g1[2] = (int)((((unsigned)K >> 16) & 0xFFFFu)
        | (((unsigned)rows_rem & 0xFFFFu) << 16));                 // dim0 hi | dim1 lo
  g1[3] = (int)((((unsigned)rows_rem >> 16) & 0xFFFFu)
        | (32u << 16));                                            // dim1 hi | tile_dim0=32
  g1[4] = 128;                                                     // tile_dim1=128 (tile_dim2=0)
  g1[5] = K;                                                       // tensor_dim0_stride lo32
  g1[6] = 0;
  g1[7] = 0;
  v4i gz4 = (v4i)0;
  v8i gz8 = (v8i)0;
  // amdgpu-toolchain (clang-23) 6-arg form: (g0, g1, g2, g3, g4, cpol)
  __builtin_amdgcn_tensor_load_to_lds(g0, g1, gz4, gz4, gz8, 0);
}
#endif

// ---------------- elementwise / prep kernels ----------------

__global__ void cvt_bf16(const float* __restrict__ in, bf16_t* __restrict__ out, int n){
  int i = blockIdx.x * blockDim.x + threadIdx.x;
  if (i < n) out[i] = (bf16_t)in[i];
}

__global__ void prep_masks(const int* __restrict__ cond, const int* __restrict__ src_mask,
                           const int* __restrict__ re_mask,
                           float* __restrict__ add_kt, float* __restrict__ mul_vt,
                           float* __restrict__ add_kr, float* __restrict__ mul_vr,
                           float* __restrict__ add_km, float* __restrict__ mul_vm){
  int i = blockIdx.x * blockDim.x + threadIdx.x;
  if (i >= Bn * Tn) return;
  int b = i >> 10, t = i & 1023;
  int ct = cond[b];
  float tct = (ct % 10 > 0) ? 1.0f : 0.0f;
  float rct = (ct / 10 > 0) ? 1.0f : 0.0f;
  float src = (float)src_mask[i];
  add_km[i] = (1.0f - src) * NEGV;
  mul_vm[i] = src;
  if (t < NTn){ int j = b * NTn + t; add_kt[j] = (1.0f - tct) * NEGV; mul_vt[j] = tct; }
  if (t < RMn){
    int j = b * RMn + t;
    float rm = (float)re_mask[j];
    add_kr[j] = (1.0f - rct) * NEGV + (1.0f - rm) * NEGV;
    mul_vr[j] = rct * rm;
  }
}

// generic LayerNorm (wave per row) -> bf16 output
__global__ __launch_bounds__(256) void ln_rows_bf16(const float* __restrict__ in,
    const float* __restrict__ g, const float* __restrict__ bb,
    bf16_t* __restrict__ out, int rows, int cols){
  int wave = threadIdx.x >> 5, lane = threadIdx.x & 31;
  int row = blockIdx.x * 8 + wave;
  if (row >= rows) return;
  const float* p = in + (size_t)row * cols;
  float s = 0.f, sq = 0.f;
  for (int c = lane; c < cols; c += 32){ float x = p[c]; s += x; sq += x * x; }
  s = wave_sum(s); sq = wave_sum(sq);
  float m = s / cols;
  float inv = rsqrtf(sq / cols - m * m + 1e-5f);
  bf16_t* o = out + (size_t)row * cols;
  for (int c = lane; c < cols; c += 32) o[c] = (bf16_t)((p[c] - m) * inv * g[c] + bb[c]);
}

// LayerNorm of concat(re_motion, re_text broadcast) gathered on the fly -> bf16
__global__ __launch_bounds__(256) void ln_rekey(const float* __restrict__ re_motion,
    const float* __restrict__ re_text, const float* __restrict__ g,
    const float* __restrict__ bb, bf16_t* __restrict__ out){
  int wave = threadIdx.x >> 5, lane = threadIdx.x & 31;
  int row = blockIdx.x * 8 + wave;             // B*RM rows
  if (row >= Bn * RMn) return;
  int b = row / RMn, rm = row % RMn;
  int r = rm / Mm, mm = rm % Mm;
  const float* mot = re_motion + ((size_t)((b * Rn + r) * Mm + mm)) * Dn;
  const float* txt = re_text + ((size_t)(b * Rn + r)) * Dn;
  const int cols = 2 * Dn;
  float s = 0.f, sq = 0.f;
  for (int c = lane; c < cols; c += 32){
    float x = (c < Dn) ? mot[c] : txt[c - Dn];
    s += x; sq += x * x;
  }
  s = wave_sum(s); sq = wave_sum(sq);
  float m = s / cols;
  float inv = rsqrtf(sq / cols - m * m + 1e-5f);
  bf16_t* o = out + (size_t)row * cols;
  for (int c = lane; c < cols; c += 32){
    float x = (c < Dn) ? mot[c] : txt[c - Dn];
    o[c] = (bf16_t)((x - m) * inv * g[c] + bb[c]);
  }
}

// ---------------- tiled bf16 WMMA GEMM: C = epilogue(A @ W + bias) ----------------
// A: (Mtot,K) bf16 row-major; W: (K,ldw) bf16 row-major; out rows remapped:
//   orow = (r / in_pb) * out_pb + row_off + (r % in_pb)
#define BM 128
#define BNt 64
#define BK 32
#define BKP 40

__global__ __launch_bounds__(256) void gemm_bf16_wmma(
    const bf16_t* __restrict__ A, const bf16_t* __restrict__ W,
    const float* __restrict__ bias, float* __restrict__ C,
    int Mtot, int K, int ldw, int ldc,
    const float* __restrict__ addv, const float* __restrict__ mulv,
    const float* __restrict__ resid, int in_pb, int out_pb, int row_off){
  __shared__ bf16_t As[BM * BKP];
  __shared__ bf16_t Bs[BNt * BKP];
  int tid = threadIdx.x;
  int lane = tid & 31, wave = tid >> 5;
  int warp_m = wave & 3, warp_n = wave >> 2;
  int m0 = blockIdx.x * BM;
  int n0 = blockIdx.y * BNt;
  int lrow = lane & 15;
  int kbA = (lane >> 4) * 8;    // A fragment K-chunk base
  int kbB = (lane >> 4) * 8;    // B fragment K-chunk base
  v8f acc[2][2] = {};

  for (int kk = 0; kk < K; kk += BK){
#ifdef USE_TDM
    // Tensor Data Mover stages the A tile (128x32, padded to 40-half rows, OOB->0)
    if (wave == 0)
      tdm_load_tile_a(A + (size_t)m0 * K + kk, (unsigned)(size_t)(void*)As, K, Mtot - m0);
#else
    {
      int ar = tid >> 1;
      int ac = (tid & 1) * 16;
      if (m0 + ar < Mtot){
        const bf16_t* src = A + (size_t)(m0 + ar) * K + kk + ac;
        *(v8bf*)&As[ar * BKP + ac]     = *(const v8bf*)src;
        *(v8bf*)&As[ar * BKP + ac + 8] = *(const v8bf*)(src + 8);
      } else {
#pragma unroll
        for (int i = 0; i < 16; i++) As[ar * BKP + ac + i] = (bf16_t)0.0f;
      }
    }
#endif
    // stage W tile transposed (n-major: Bs[n][k]) for 16B fragment loads
    {
      int bnq = (tid & 7) * 8;
      int bkq = tid >> 3;
      v8bf wv = *(const v8bf*)&W[(size_t)(kk + bkq) * ldw + n0 + bnq];
#pragma unroll
      for (int i = 0; i < 8; i++) Bs[(bnq + i) * BKP + bkq] = wv[i];
    }
    if (kk + BK < K){
      __builtin_prefetch(W + (size_t)(kk + BK + (tid >> 3)) * ldw + n0, 0, 3);
    }
#ifdef USE_TDM
    if (wave == 0) __builtin_amdgcn_s_wait_tensorcnt(0);
#endif
    __syncthreads();
#pragma unroll
    for (int rt = 0; rt < 2; rt++){
      FragAB af;
      int arow = warp_m * 32 + rt * 16 + lrow;
      af.p[0] = *(const v8bf*)&As[arow * BKP + kbA];
      af.p[1] = *(const v8bf*)&As[arow * BKP + kbA + 16];
#pragma unroll
      for (int ctl = 0; ctl < 2; ctl++){
        FragAB bfv;
        int brow = warp_n * 32 + ctl * 16 + lrow;
        bfv.p[0] = *(const v8bf*)&Bs[brow * BKP + kbB];
        bfv.p[1] = *(const v8bf*)&Bs[brow * BKP + kbB + 16];
        acc[rt][ctl] = __builtin_amdgcn_wmma_f32_16x16x32_bf16(
            false, af.v, false, bfv.v, (short)0, acc[rt][ctl], false, false);
      }
    }
    __syncthreads();
  }
  int hi8 = (lane >> 4) * 8;
#pragma unroll
  for (int rt = 0; rt < 2; rt++)
#pragma unroll
    for (int ctl = 0; ctl < 2; ctl++)
#pragma unroll
      for (int j = 0; j < 8; j++){
        int r = m0 + warp_m * 32 + rt * 16 + hi8 + j;
        int c = n0 + warp_n * 32 + ctl * 16 + lrow;
        if (r < Mtot){
          float v = acc[rt][ctl][j] + bias[c];
          if (addv) v += addv[r];
          if (mulv) v *= mulv[r];
          int orow = (r / in_pb) * out_pb + row_off + (r % in_pb);
          size_t oidx = (size_t)orow * ldc + c;
          if (resid) v += resid[oidx];
          C[oidx] = v;
        }
      }
}

// ---------------- softmaxes + transposes ----------------

// softmax over d_head (64) per (b,t,h); write bf16 in same layout
__global__ void softmax_q(const float* __restrict__ qb, bf16_t* __restrict__ qo){
  int gid = blockIdx.x * blockDim.x + threadIdx.x;
  if (gid >= Bn * Tn * Hn) return;
  const float* p = qb + (size_t)gid * DHn;
  bf16_t* o = qo + (size_t)gid * DHn;
  float mx = -3.4e38f;
  for (int i = 0; i < DHn; i++) mx = fmaxf(mx, p[i]);
  float s = 0.f;
  for (int i = 0; i < DHn; i++) s += __expf(p[i] - mx);
  float inv = 1.0f / s;
  for (int i = 0; i < DHn; i++) o[i] = (bf16_t)(__expf(p[i] - mx) * inv);
}

// softmax over N per (b, d_full); writes transposed bf16 kT[b][d][n], zero-padded to NPAD
__global__ void softmax_k_t(const float* __restrict__ key, bf16_t* __restrict__ kT){
  int gid = blockIdx.x * blockDim.x + threadIdx.x;
  if (gid >= Bn * Dn) return;
  int b = gid / Dn, d = gid % Dn;
  const float* p = key + (size_t)b * Nn * Dn + d;
  float mx = -3.4e38f;
  for (int n = 0; n < Nn; n++) mx = fmaxf(mx, p[(size_t)n * Dn]);
  float s = 0.f;
  for (int n = 0; n < Nn; n++) s += __expf(p[(size_t)n * Dn] - mx);
  float inv = 1.0f / s;
  bf16_t* o = kT + ((size_t)b * Dn + d) * NPAD;
  for (int n = 0; n < Nn; n++) o[n] = (bf16_t)(__expf(p[(size_t)n * Dn] - mx) * inv);
  for (int n = Nn; n < NPAD; n++) o[n] = (bf16_t)0.0f;
}

__global__ void transpose_v(const float* __restrict__ v, bf16_t* __restrict__ vT){
  size_t gid = (size_t)blockIdx.x * blockDim.x + threadIdx.x;
  size_t tot = (size_t)Bn * Dn * NPAD;
  if (gid >= tot) return;
  int n = (int)(gid % NPAD);
  size_t rem = gid / NPAD;
  int d = (int)(rem % Dn);
  int b = (int)(rem / Dn);
  float val = (n < Nn) ? v[((size_t)b * Nn + n) * Dn + d] : 0.0f;
  vT[gid] = (bf16_t)val;
}

// ---------------- attention = K^T @ V per (b,h): 64x64, K-dim = NPAD ----------------
// output written transposed bf16: attnT[(b*H+h)][l][d]
__global__ __launch_bounds__(128) void attn_gemm(const bf16_t* __restrict__ kT,
    const bf16_t* __restrict__ vT, bf16_t* __restrict__ attnT){
  int lane = threadIdx.x & 31, wave = threadIdx.x >> 5;
  int bh = blockIdx.x;
  int h = bh & 7, b = bh >> 3;
  int lrow = lane & 15;
  int kbA = (lane >> 4) * 8, kbB = (lane >> 4) * 8;
  v8f acc[4] = {};
  const bf16_t* ab = kT + ((size_t)b * Dn + h * DHn + wave * 16 + lrow) * NPAD;
  for (int kk = 0; kk < NPAD; kk += 32){
    FragAB af;
    af.p[0] = *(const v8bf*)(ab + kk + kbA);
    af.p[1] = *(const v8bf*)(ab + kk + kbA + 16);
#pragma unroll
    for (int ctl = 0; ctl < 4; ctl++){
      FragAB bfv;
      const bf16_t* bp = vT + ((size_t)b * Dn + h * DHn + ctl * 16 + lrow) * NPAD + kk + kbB;
      bfv.p[0] = *(const v8bf*)bp;
      bfv.p[1] = *(const v8bf*)(bp + 16);
      acc[ctl] = __builtin_amdgcn_wmma_f32_16x16x32_bf16(
          false, af.v, false, bfv.v, (short)0, acc[ctl], false, false);
    }
  }
  int hi8 = (lane >> 4) * 8;
#pragma unroll
  for (int ctl = 0; ctl < 4; ctl++)
#pragma unroll
    for (int j = 0; j < 8; j++){
      int d = wave * 16 + hi8 + j;
      int l = ctl * 16 + lrow;
      attnT[((size_t)bh * 64 + l) * 64 + d] = (bf16_t)acc[ctl][j];
    }
}

// ---------------- y = q @ attn per (b,h): (T x 64) @ (64 x 64) ----------------
__global__ __launch_bounds__(128) void y_headgemm(const bf16_t* __restrict__ qbf,
    const bf16_t* __restrict__ attnT, float* __restrict__ y){
  int lane = threadIdx.x & 31, wave = threadIdx.x >> 5;
  int blk = blockIdx.x;            // B*H*(T/64)
  int mt = blk & 15;
  int bh = blk >> 4;
  int h = bh & 7, b = bh >> 3;
  int t0 = mt * 64 + wave * 16;
  int lrow = lane & 15;
  int kbA = (lane >> 4) * 8, kbB = (lane >> 4) * 8;
  v8f acc[4] = {};
  const bf16_t* ab = qbf + ((size_t)(b * Tn + t0 + lrow)) * Dn + h * DHn;
  const bf16_t* bb = attnT + (size_t)bh * 64 * 64;
#pragma unroll
  for (int kk = 0; kk < DHn; kk += 32){
    FragAB af;
    af.p[0] = *(const v8bf*)(ab + kk + kbA);
    af.p[1] = *(const v8bf*)(ab + kk + kbA + 16);
#pragma unroll
    for (int ctl = 0; ctl < 4; ctl++){
      FragAB bfv;
      const bf16_t* bp = bb + (ctl * 16 + lrow) * 64 + kk + kbB;
      bfv.p[0] = *(const v8bf*)bp;
      bfv.p[1] = *(const v8bf*)(bp + 16);
      acc[ctl] = __builtin_amdgcn_wmma_f32_16x16x32_bf16(
          false, af.v, false, bfv.v, (short)0, acc[ctl], false, false);
    }
  }
  int hi8 = (lane >> 4) * 8;
#pragma unroll
  for (int ctl = 0; ctl < 4; ctl++)
#pragma unroll
    for (int j = 0; j < 8; j++){
      int t = t0 + hi8 + j;
      int c = ctl * 16 + lrow;
      y[((size_t)(b * Tn + t)) * Dn + h * DHn + c] = acc[ctl][j];
    }
}

// ---------------- stylization ----------------

__global__ void silu_emb(const float* __restrict__ emb, float* __restrict__ out, int n){
  int i = blockIdx.x * blockDim.x + threadIdx.x;
  if (i < n) out[i] = silu_f(emb[i]);
}

__global__ void emb_gemm(const float* __restrict__ semb, const float* __restrict__ W,
                         const float* __restrict__ bias, float* __restrict__ out){
  int gid = blockIdx.x * blockDim.x + threadIdx.x;
  if (gid >= Bn * 2 * Dn) return;
  int b = gid >> 10, j = gid & 1023;
  const float* e = semb + (size_t)b * En;
  float s = bias[j];
  for (int k = 0; k < En; k++) s += e[k] * W[(size_t)k * (2 * Dn) + j];
  out[gid] = s;
}

// hmod = silu(LN(y)*(1+scale)+shift) as bf16
__global__ __launch_bounds__(256) void modulate(const float* __restrict__ y,
    const float* __restrict__ g, const float* __restrict__ bb,
    const float* __restrict__ embo, bf16_t* __restrict__ hmod){
  int wave = threadIdx.x >> 5, lane = threadIdx.x & 31;
  int row = blockIdx.x * 8 + wave;
  if (row >= Bn * Tn) return;
  int b = row >> 10;
  const float* p = y + (size_t)row * Dn;
  float s = 0.f, sq = 0.f;
  for (int c = lane; c < Dn; c += 32){ float x = p[c]; s += x; sq += x * x; }
  s = wave_sum(s); sq = wave_sum(sq);
  float m = s / Dn;
  float inv = rsqrtf(sq / Dn - m * m + 1e-5f);
  const float* eo = embo + (size_t)b * (2 * Dn);
  bf16_t* o = hmod + (size_t)row * Dn;
  for (int c = lane; c < Dn; c += 32){
    float u = (p[c] - m) * inv * g[c] + bb[c];
    u = u * (1.0f + eo[c]) + eo[Dn + c];
    o[c] = (bf16_t)silu_f(u);
  }
}

// ---------------- host orchestration ----------------

extern "C" void kernel_launch(void* const* d_in, const int* in_sizes, int n_in,
                              void* d_out, int out_size, void* d_ws, size_t ws_size,
                              hipStream_t stream) {
  (void)in_sizes; (void)n_in; (void)out_size; (void)ws_size;
  const float* x         = (const float*)d_in[0];
  const float* xf        = (const float*)d_in[1];
  const float* emb       = (const float*)d_in[2];
  const float* re_motion = (const float*)d_in[3];
  const float* re_text   = (const float*)d_in[4];
  const int*   src_mask  = (const int*)d_in[5];
  const int*   re_mask   = (const int*)d_in[6];
  const int*   cond_type = (const int*)d_in[7];
  const float* norm_g = (const float*)d_in[8],  *norm_b = (const float*)d_in[9];
  const float* tnorm_g = (const float*)d_in[10], *tnorm_b = (const float*)d_in[11];
  const float* rn1_g = (const float*)d_in[12],  *rn1_b = (const float*)d_in[13];
  const float* rn2_g = (const float*)d_in[14],  *rn2_b = (const float*)d_in[15];
  const float* snorm_g = (const float*)d_in[16], *snorm_b = (const float*)d_in[17];
  const float* Wq  = (const float*)d_in[18], *bq  = (const float*)d_in[19];
  const float* Wkt = (const float*)d_in[20], *bkt = (const float*)d_in[21];
  const float* Wvt = (const float*)d_in[22], *bvt = (const float*)d_in[23];
  const float* Wkm = (const float*)d_in[24], *bkm = (const float*)d_in[25];
  const float* Wvm = (const float*)d_in[26], *bvm = (const float*)d_in[27];
  const float* Wkr = (const float*)d_in[28], *bkr = (const float*)d_in[29];
  const float* Wvr = (const float*)d_in[30], *bvr = (const float*)d_in[31];
  const float* emb_W = (const float*)d_in[32], *emb_b = (const float*)d_in[33];
  const float* out_W = (const float*)d_in[34], *out_b = (const float*)d_in[35];
  float* out = (float*)d_out;

  char* base = (char*)d_ws;
  size_t off = 0;
  auto alloc = [&](size_t bytes) -> void* {
    void* p = base + off;
    off += (bytes + 255) & ~(size_t)255;
    return p;
  };

  bf16_t* xn_bf   = (bf16_t*)alloc((size_t)Bn * Tn * Dn * 2);
  bf16_t* xfn_bf  = (bf16_t*)alloc((size_t)Bn * NTn * DTn * 2);
  bf16_t* rekey_bf = (bf16_t*)alloc((size_t)Bn * RMn * 2 * Dn * 2);
  bf16_t* reval_bf = (bf16_t*)alloc((size_t)Bn * RMn * Dn * 2);
  bf16_t* Wq_bf  = (bf16_t*)alloc((size_t)Dn * Dn * 2);
  bf16_t* Wkt_bf = (bf16_t*)alloc((size_t)DTn * Dn * 2);
  bf16_t* Wvt_bf = (bf16_t*)alloc((size_t)DTn * Dn * 2);
  bf16_t* Wkm_bf = (bf16_t*)alloc((size_t)Dn * Dn * 2);
  bf16_t* Wvm_bf = (bf16_t*)alloc((size_t)Dn * Dn * 2);
  bf16_t* Wkr_bf = (bf16_t*)alloc((size_t)2 * Dn * Dn * 2);
  bf16_t* Wvr_bf = (bf16_t*)alloc((size_t)Dn * Dn * 2);
  bf16_t* Wo_bf  = (bf16_t*)alloc((size_t)Dn * Dn * 2);
  float*  q_buf  = (float*)alloc((size_t)Bn * Tn * Dn * 4);   // reused as y later
  bf16_t* q_bf   = (bf16_t*)alloc((size_t)Bn * Tn * Dn * 2);
  float*  keyb   = (float*)alloc((size_t)Bn * Nn * Dn * 4);
  float*  valb   = (float*)alloc((size_t)Bn * Nn * Dn * 4);
  bf16_t* kT     = (bf16_t*)alloc((size_t)Bn * Dn * NPAD * 2);
  bf16_t* vT     = (bf16_t*)alloc((size_t)Bn * Dn * NPAD * 2);
  bf16_t* attnT  = (bf16_t*)alloc((size_t)Bn * Hn * 64 * 64 * 2);
  float*  semb   = (float*)alloc((size_t)Bn * En * 4);
  float*  embo   = (float*)alloc((size_t)Bn * 2 * Dn * 4);
  bf16_t* hmod   = (bf16_t*)alloc((size_t)Bn * Tn * Dn * 2);
  float* add_kt = (float*)alloc((size_t)Bn * NTn * 4);
  float* mul_vt = (float*)alloc((size_t)Bn * NTn * 4);
  float* add_kr = (float*)alloc((size_t)Bn * RMn * 4);
  float* mul_vr = (float*)alloc((size_t)Bn * RMn * 4);
  float* add_km = (float*)alloc((size_t)Bn * Tn * 4);
  float* mul_vm = (float*)alloc((size_t)Bn * Tn * 4);
  float* ybuf = q_buf;  // alias: q_buf consumed by softmax_q before y is written

  auto cvt = [&](const float* in, bf16_t* o, int n){
    cvt_bf16<<<(n + 255) / 256, 256, 0, stream>>>(in, o, n);
  };
  cvt(Wq, Wq_bf, Dn * Dn);      cvt(Wkt, Wkt_bf, DTn * Dn);
  cvt(Wvt, Wvt_bf, DTn * Dn);   cvt(Wkm, Wkm_bf, Dn * Dn);
  cvt(Wvm, Wvm_bf, Dn * Dn);    cvt(Wkr, Wkr_bf, 2 * Dn * Dn);
  cvt(Wvr, Wvr_bf, Dn * Dn);    cvt(out_W, Wo_bf, Dn * Dn);

  prep_masks<<<(Bn * Tn + 255) / 256, 256, 0, stream>>>(
      cond_type, src_mask, re_mask, add_kt, mul_vt, add_kr, mul_vr, add_km, mul_vm);

  ln_rows_bf16<<<(Bn * Tn) / 8, 256, 0, stream>>>(x, norm_g, norm_b, xn_bf, Bn * Tn, Dn);
  ln_rows_bf16<<<(Bn * NTn) / 8, 256, 0, stream>>>(xf, tnorm_g, tnorm_b, xfn_bf, Bn * NTn, DTn);
  ln_rekey<<<(Bn * RMn) / 8, 256, 0, stream>>>(re_motion, re_text, rn1_g, rn1_b, rekey_bf);
  ln_rows_bf16<<<(Bn * RMn) / 8, 256, 0, stream>>>(re_motion, rn2_g, rn2_b, reval_bf, Bn * RMn, Dn);

  auto gemm = [&](const bf16_t* A, const bf16_t* W, const float* bias, float* C,
                  int Mtot, int K, const float* addv, const float* mulv,
                  const float* resid, int in_pb, int out_pb, int row_off){
    dim3 grid((Mtot + BM - 1) / BM, Dn / BNt);
    gemm_bf16_wmma<<<grid, 256, 0, stream>>>(A, W, bias, C, Mtot, K, Dn, Dn,
                                             addv, mulv, resid, in_pb, out_pb, row_off);
  };

  // query -> q_buf (identity map)
  gemm(xn_bf, Wq_bf, bq, q_buf, Bn * Tn, Dn, nullptr, nullptr, nullptr, Bn * Tn, Bn * Tn, 0);
  // keys -> keyb (B, 1885, 512)
  gemm(xfn_bf,  Wkt_bf, bkt, keyb, Bn * NTn, DTn,    add_kt, nullptr, nullptr, NTn, Nn, 0);
  gemm(rekey_bf, Wkr_bf, bkr, keyb, Bn * RMn, 2 * Dn, add_kr, nullptr, nullptr, RMn, Nn, NTn);
  gemm(xn_bf,   Wkm_bf, bkm, keyb, Bn * Tn,  Dn,     add_km, nullptr, nullptr, Tn, Nn, NTn + RMn);
  // values -> valb
  gemm(xfn_bf,  Wvt_bf, bvt, valb, Bn * NTn, DTn, nullptr, mul_vt, nullptr, NTn, Nn, 0);
  gemm(reval_bf, Wvr_bf, bvr, valb, Bn * RMn, Dn, nullptr, mul_vr, nullptr, RMn, Nn, NTn);
  gemm(xn_bf,   Wvm_bf, bvm, valb, Bn * Tn,  Dn, nullptr, mul_vm, nullptr, Tn, Nn, NTn + RMn);

  softmax_q<<<(Bn * Tn * Hn + 255) / 256, 256, 0, stream>>>(q_buf, q_bf);
  softmax_k_t<<<(Bn * Dn + 255) / 256, 256, 0, stream>>>(keyb, kT);
  {
    size_t tot = (size_t)Bn * Dn * NPAD;
    transpose_v<<<(unsigned)((tot + 255) / 256), 256, 0, stream>>>(valb, vT);
  }

  attn_gemm<<<Bn * Hn, 128, 0, stream>>>(kT, vT, attnT);
  y_headgemm<<<Bn * Hn * (Tn / 64), 128, 0, stream>>>(q_bf, attnT, ybuf);

  silu_emb<<<(Bn * En + 255) / 256, 256, 0, stream>>>(emb, semb, Bn * En);
  emb_gemm<<<(Bn * 2 * Dn + 255) / 256, 256, 0, stream>>>(semb, emb_W, emb_b, embo);
  modulate<<<(Bn * Tn) / 8, 256, 0, stream>>>(ybuf, snorm_g, snorm_b, embo, hmod);

  // final: out = x + (hmod @ out_W + out_b)
  gemm(hmod, Wo_bf, out_b, out, Bn * Tn, Dn, nullptr, nullptr, x, Bn * Tn, Bn * Tn, 0);
}
